// GraphEncoder_51058571215016
// MI455X (gfx1250) — matrix operand order repaired and checked by reading, not compiled, not run on hardware
//
#include <hip/hip_runtime.h>
#include <hip/hip_bf16.h>
#include <string.h>

// ---------------- problem constants ----------------
constexpr int NG   = 512;          // graphs
constexpr int NP   = 64;           // nodes per graph
constexpr int NN   = NG * NP;      // 32768 nodes
constexpr int DEG  = 8;
constexpr int EPG  = NP * DEG;     // 512 edges per graph
constexpr int NE   = NN * DEG;     // 262144 edges
constexpr int PEK  = 6;
constexpr int IND  = 128;
constexpr int HID  = 256;
constexpr int EDIM = 32;
constexpr int KINIT = 160;         // 128 + 6 padded to multiple of 32
constexpr float EPS_BN = 1e-5f;

constexpr int ASTR = 40;           // A-tile LDS stride (ushorts): 80B, 16B aligned

// ---------------- WMMA types ----------------
typedef __attribute__((ext_vector_type(16))) __bf16 v16bf;
typedef __attribute__((ext_vector_type(8)))  float  v8f;

union BF16x16 { v16bf v; unsigned short u[16]; uint4 q[2]; };

__device__ __forceinline__ unsigned short f2bf(float x) {
  unsigned int u = __builtin_bit_cast(unsigned int, x);
  unsigned int r = u + 0x7FFFu + ((u >> 16) & 1u);   // round-to-nearest-even
  return (unsigned short)(r >> 16);
}

// =====================================================================
// K0: per-graph normalized-Laplacian eigensolver (parallel Jacobi)
// one 64-thread block per graph; A,V in LDS
// =====================================================================
__global__ void lap_pe_kernel(const int* __restrict__ edge_index,
                              float* __restrict__ pos) {
  extern __shared__ char smem[];
  float* Am   = (float*)smem;           // 64x64
  float* Vm   = Am + 64 * 64;           // 64x64
  float* dinv = Vm + 64 * 64;           // 64
  float* ev   = dinv + 64;              // 64
  float* cc   = ev + 64;                // 32
  float* ss   = cc + 32;                // 32
  int*   pp   = (int*)(ss + 32);        // 32
  int*   qq   = pp + 32;                // 32
  int*   idxs = qq + 32;                // 64

  const int g = blockIdx.x;
  const int t = threadIdx.x;            // 0..63

  for (int i = t; i < 64 * 64; i += 64) { Am[i] = 0.f; Vm[i] = 0.f; }
  __syncthreads();

  // adjacency with multiplicity (exact integer fp adds -> deterministic)
  for (int e = t; e < EPG; e += 64) {
    int s = edge_index[g * EPG + e] & (NP - 1);
    int d = edge_index[NE + g * EPG + e] & (NP - 1);
    atomicAdd(&Am[s * 64 + d], 1.0f);
  }
  __syncthreads();

  // out-degree and D^{-1/2}
  {
    float dsum = 0.f;
    for (int j = 0; j < 64; ++j) dsum += Am[t * 64 + j];
    dinv[t] = dsum > 0.f ? rsqrtf(dsum) : 0.f;
  }
  __syncthreads();

  // L = I - dinv_i * A_ij * dinv_j  (row t)
  for (int j = 0; j < 64; ++j)
    Am[t * 64 + j] = (t == j ? 1.f : 0.f) - dinv[t] * Am[t * 64 + j] * dinv[j];
  __syncthreads();
  // symmetrize from lower triangle (eigh UPLO='L' convention)
  for (int j = t + 1; j < 64; ++j) Am[t * 64 + j] = Am[j * 64 + t];
  __syncthreads();
  for (int j = 0; j < 64; ++j) Vm[t * 64 + j] = (t == j) ? 1.f : 0.f;
  __syncthreads();

  // parallel cyclic Jacobi: round-robin tournament pairing, 32 disjoint pairs
  for (int sweep = 0; sweep < 12; ++sweep) {
    for (int r = 0; r < 63; ++r) {
      if (t < 32) {
        int p, q;
        if (t == 0) { p = 63; q = r % 63; }
        else { p = (r + t) % 63; q = (r + 63 - t) % 63; }
        float app = Am[p * 64 + p], aqq = Am[q * 64 + q], apq = Am[p * 64 + q];
        float c = 1.f, s = 0.f;
        if (fabsf(apq) > 1e-12f) {
          float tau = (aqq - app) / (2.f * apq);
          float tt = (tau >= 0.f ? 1.f : -1.f) / (fabsf(tau) + sqrtf(1.f + tau * tau));
          c = rsqrtf(1.f + tt * tt);
          s = tt * c;
        }
        pp[t] = p; qq[t] = q; cc[t] = c; ss[t] = s;
      }
      __syncthreads();
      // row phase: A <- J^T A   (thread t owns column t)
      for (int k = 0; k < 32; ++k) {
        int p = pp[k], q = qq[k]; float c = cc[k], s = ss[k];
        float ap = Am[p * 64 + t], aq = Am[q * 64 + t];
        Am[p * 64 + t] = c * ap - s * aq;
        Am[q * 64 + t] = s * ap + c * aq;
      }
      __syncthreads();
      // column phase: A <- A J, V <- V J  (thread t owns row t)
      for (int k = 0; k < 32; ++k) {
        int p = pp[k], q = qq[k]; float c = cc[k], s = ss[k];
        float ap = Am[t * 64 + p], aq = Am[t * 64 + q];
        Am[t * 64 + p] = c * ap - s * aq;
        Am[t * 64 + q] = s * ap + c * aq;
        float vp = Vm[t * 64 + p], vq = Vm[t * 64 + q];
        Vm[t * 64 + p] = c * vp - s * vq;
        Vm[t * 64 + q] = s * vp + c * vq;
      }
      __syncthreads();
    }
  }

  ev[t] = Am[t * 64 + t];
  __syncthreads();
  if (t == 0) {
    for (int i = 0; i < 64; ++i) idxs[i] = i;
    for (int a = 0; a < PEK; ++a) {           // partial selection sort, ascending
      int mn = a;
      for (int b = a + 1; b < 64; ++b)
        if (ev[idxs[b]] < ev[idxs[mn]]) mn = b;
      int tmp = idxs[a]; idxs[a] = idxs[mn]; idxs[mn] = tmp;
    }
  }
  __syncthreads();
  for (int k = 0; k < PEK; ++k)
    pos[(size_t)(g * NP + t) * PEK + k] = Vm[t * 64 + idxs[k]];
}

// =====================================================================
// weight -> bf16 converters (B stored transposed: [n][k], k contiguous)
// =====================================================================
__global__ void cvt_init_kernel(const float* __restrict__ Winit,
                                unsigned short* __restrict__ out) {
  int idx = blockIdx.x * blockDim.x + threadIdx.x;  // 256*160
  if (idx >= HID * KINIT) return;
  int n = idx / KINIT, k = idx % KINIT;
  float v = (k < IND + PEK) ? Winit[(size_t)k * HID + n] : 0.f;
  out[idx] = f2bf(v);
}

__global__ void cvt_wc_kernel(const float* __restrict__ Wc1,
                              const float* __restrict__ Wc2,
                              unsigned short* __restrict__ out) {
  int idx = blockIdx.x * blockDim.x + threadIdx.x;  // 8*256*256
  if (idx >= 8 * HID * HID) return;
  int which = idx >> 16; int rem = idx & 65535;
  int n = rem >> 8, k = rem & 255;                  // out[which][n][k]
  float v;
  if (which < 4) v = Wc1[((size_t)(which * HID + n) * HID + k) * 3 + 1];
  else           v = Wc2[((size_t)((which - 4) * HID + n) * HID + k) * 3 + 1];
  out[idx] = f2bf(v);
}

__global__ void cvt_wo_kernel(const float* __restrict__ Wo,
                              unsigned short* __restrict__ out) {
  int idx = blockIdx.x * blockDim.x + threadIdx.x;  // 256*256
  if (idx >= HID * HID) return;
  int n = idx >> 8, k = idx & 255;
  out[idx] = f2bf(Wo[(size_t)k * HID + n]);
}

// =====================================================================
// WMMA GEMM: C[M x 256] = act( A[M x K] * B[K x 256] + bias (+resid) )
// B bf16, LDS-resident, stored transposed [256][Kp].
// 256 threads = 8 waves; block tile 64 rows x 256 cols; M % 64 == 0.
// wave tile 32 rows x 64 cols: 8x v_wmma_f32_16x16x32_bf16 per K-step,
// B fragments reused across the two 16-row subtiles.
// =====================================================================
__device__ __forceinline__ float gemm_loadA_slow(const float* A, int lda, int kaCols,
                                                 const float* A2, int r, int k) {
  if (k < kaCols) return A[(size_t)r * lda + k];
  if (A2 != nullptr && k < kaCols + PEK) return A2[(size_t)r * PEK + (k - kaCols)];
  return 0.f;
}

__global__ void wmma_gemm256_kernel(const float* __restrict__ A, int lda, int kaCols,
                                    const float* __restrict__ A2,
                                    const unsigned short* __restrict__ Bt, int Kp,
                                    const float* __restrict__ bias,
                                    const float* __restrict__ resid,
                                    float* __restrict__ C, int act) {
  extern __shared__ char smem[];
  unsigned short* Bs = (unsigned short*)smem;                       // [256][Kp]
  unsigned short* As = (unsigned short*)(smem + (size_t)512 * Kp);  // [64][ASTR]

  const int tid  = threadIdx.x;
  const int lane = tid & 31;
  const int wave = tid >> 5;
  const int wm   = wave >> 2;        // 0..1  (32-row half of block)
  const int wn   = wave & 3;         // 0..3  (64-col strip)
  const int half = lane >> 4;
  const int l15  = lane & 15;
  const int rowBase = blockIdx.x * 64;
  const bool fastA = (A2 == nullptr) && (kaCols >= Kp);

  // stage full bf16 weight matrix into LDS (<=128KB)
  {
    const uint4* src = (const uint4*)Bt;
    uint4* dst = (uint4*)Bs;
    const int chunks = 32 * Kp;        // 256*Kp*2/16
    for (int i = tid; i < chunks; i += 256) dst[i] = src[i];
  }

  v8f acc[2][4];
#pragma unroll
  for (int rt = 0; rt < 2; ++rt)
#pragma unroll
    for (int ct = 0; ct < 4; ++ct) {
      v8f z = {0.f, 0.f, 0.f, 0.f, 0.f, 0.f, 0.f, 0.f};
      acc[rt][ct] = z;
    }

  const int arow0 = tid >> 2;          // 0..63  (A stage row)
  const int akq   = (tid & 3) * 8;     // 0,8,16,24

  for (int k0 = 0; k0 < Kp; k0 += 32) {
    // ---- stage 64x32 f32->bf16 A tile (8 consecutive k per thread) ----
    {
      unsigned short tmp[8];
      if (fastA) {
        const float4* src = (const float4*)(A + (size_t)(rowBase + arow0) * lda + k0 + akq);
        float4 a0 = src[0], a1 = src[1];
        tmp[0] = f2bf(a0.x); tmp[1] = f2bf(a0.y); tmp[2] = f2bf(a0.z); tmp[3] = f2bf(a0.w);
        tmp[4] = f2bf(a1.x); tmp[5] = f2bf(a1.y); tmp[6] = f2bf(a1.z); tmp[7] = f2bf(a1.w);
      } else {
#pragma unroll
        for (int i = 0; i < 8; ++i)
          tmp[i] = f2bf(gemm_loadA_slow(A, lda, kaCols, A2, rowBase + arow0, k0 + akq + i));
      }
      *(uint4*)&As[arow0 * ASTR + akq] = *(const uint4*)tmp;  // 16B aligned
    }
    __syncthreads();

    // ---- A fragments (contiguous k = 16*half + j, matching B) ----
    BF16x16 af[2];
#pragma unroll
    for (int rt = 0; rt < 2; ++rt) {
      int arow = wm * 32 + rt * 16 + l15;
#pragma unroll
      for (int j = 0; j < 16; ++j) {
        int kk = j + 16 * half;
        af[rt].u[j] = As[arow * ASTR + kk];
      }
    }

#pragma unroll
    for (int ct = 0; ct < 4; ++ct) {
      BF16x16 bf;
      int ncol = wn * 64 + ct * 16 + l15;
      const uint4* bp = (const uint4*)&Bs[(size_t)ncol * Kp + k0 + half * 16];
      bf.q[0] = bp[0];
      bf.q[1] = bp[1];
#if defined(__gfx1250__)
      acc[0][ct] = __builtin_amdgcn_wmma_f32_16x16x32_bf16(
          false, af[0].v, false, bf.v, (short)0, acc[0][ct], false, false);
      acc[1][ct] = __builtin_amdgcn_wmma_f32_16x16x32_bf16(
          false, af[1].v, false, bf.v, (short)0, acc[1][ct], false, false);
#else
      (void)bf;
#endif
    }
    __syncthreads();
  }

  // ---- epilogue: bias (+resid) + activation ----
#pragma unroll
  for (int ct = 0; ct < 4; ++ct) {
    int col = wn * 64 + ct * 16 + l15;
    float bv = bias ? bias[col] : 0.f;
#pragma unroll
    for (int rt = 0; rt < 2; ++rt) {
#pragma unroll
      for (int rr = 0; rr < 8; ++rr) {
        int row = rowBase + wm * 32 + rt * 16 + rr + 8 * half;
        float v = acc[rt][ct][rr] + bv;
        if (resid) v += resid[(size_t)row * HID + col];
        if (act == 1)      v = v > 0.f ? v : 0.f;
        else if (act == 2) v = v > 0.f ? v : 0.02f * v;
        C[(size_t)row * HID + col] = v;
      }
    }
  }
}

// =====================================================================
// BatchNorm statistics: one block per column, fixed-order tree reduce
// =====================================================================
__global__ void bn_stats_kernel(const float* __restrict__ X, int rows,
                                float* __restrict__ mu, float* __restrict__ rs) {
  __shared__ float s1[256], s2[256];
  const int col = blockIdx.x;
  const int tid = threadIdx.x;
  float a = 0.f, b = 0.f;
  for (int r = tid; r < rows; r += 256) {
    float v = X[(size_t)r * HID + col];
    a += v; b += v * v;
  }
  s1[tid] = a; s2[tid] = b;
  __syncthreads();
  for (int st = 128; st > 0; st >>= 1) {
    if (tid < st) { s1[tid] += s1[tid + st]; s2[tid] += s2[tid + st]; }
    __syncthreads();
  }
  if (tid == 0) {
    float m = s1[0] / (float)rows;
    float var = fmaxf(s2[0] / (float)rows - m * m, 0.f);
    mu[col] = m;
    rs[col] = rsqrtf(var + EPS_BN);
  }
}

__global__ void bn_norm_kernel(const float* __restrict__ X,
                               const float* __restrict__ mu,
                               const float* __restrict__ rs,
                               const float* __restrict__ g,
                               const float* __restrict__ b,
                               float* __restrict__ Y, int total, int act) {
  int i = blockIdx.x * blockDim.x + threadIdx.x;
  if (i >= total) return;
  int col = i & (HID - 1);
  float v = (X[i] - mu[col]) * rs[col] * g[col] + b[col];
  if (act == 2) v = v > 0.f ? v : 0.02f * v;
  Y[i] = v;
}

// =====================================================================
// GINE aggregation, one block per graph:
// z = (1+eps)*h + sum_{e: dst=i} relu(h[src_e] + edge_attr_e @ We + be)
// deterministic per-dst bucket ordering (no float atomics)
// =====================================================================
__global__ void gine_agg_kernel(const float* __restrict__ h,
                                const float* __restrict__ edge_attr,
                                const int* __restrict__ edge_index,
                                const float* __restrict__ We,   // [32][256]
                                const float* __restrict__ be,   // [256]
                                const float* __restrict__ epsp, // &eps_l[l]
                                float* __restrict__ z) {
  extern __shared__ char smem[];
  float* hs  = (float*)smem;            // 64*256
  float* eas = hs + 64 * HID;           // 512*32
  float* Wes = eas + EPG * EDIM;        // 32*256
  float* bes = Wes + EDIM * HID;        // 256
  int* sl    = (int*)(bes + HID);       // 512
  int* dl    = sl + EPG;                // 512
  int* cnt   = dl + EPG;                // 64
  int* off   = cnt + 64;                // 65
  int* order = off + 65;                // 512

  const int g = blockIdx.x;
  const int tid = threadIdx.x;

  // stage h tile, edge_attr tile, We, be
  {
    const float4* src = (const float4*)(h + (size_t)g * NP * HID);
    float4* dst = (float4*)hs;
    for (int i = tid; i < NP * HID / 4; i += 256) dst[i] = src[i];
  }
  {
    const float4* src = (const float4*)(edge_attr + (size_t)g * EPG * EDIM);
    float4* dst = (float4*)eas;
    for (int i = tid; i < EPG * EDIM / 4; i += 256) dst[i] = src[i];
  }
  {
    const float4* src = (const float4*)We;
    float4* dst = (float4*)Wes;
    for (int i = tid; i < EDIM * HID / 4; i += 256) dst[i] = src[i];
  }
  if (tid < HID) bes[tid] = be[tid];
  for (int e = tid; e < EPG; e += 256) {
    sl[e] = edge_index[g * EPG + e] & (NP - 1);
    dl[e] = edge_index[NE + g * EPG + e] & (NP - 1);
  }
  __syncthreads();

  // deterministic bucket-by-dst
  if (tid < NP) {
    int c = 0;
    for (int e = 0; e < EPG; ++e) if (dl[e] == tid) ++c;
    cnt[tid] = c;
  }
  __syncthreads();
  if (tid == 0) {
    int run = 0;
    for (int i = 0; i < NP; ++i) { off[i] = run; run += cnt[i]; }
    off[NP] = run;
  }
  __syncthreads();
  if (tid < NP) {
    int o = off[tid];
    for (int e = 0; e < EPG; ++e) if (dl[e] == tid) order[o++] = e;
  }
  __syncthreads();

  // compute: thread = (node, 64-feature chunk)
  const int node  = tid >> 2;
  const int fbase = (tid & 3) * 64;
  const float epsv = 1.0f + epsp[0];

  float acc[64];
#pragma unroll
  for (int f = 0; f < 64; ++f) acc[f] = epsv * hs[node * HID + fbase + f];

  const int o0 = off[node], o1 = off[node + 1];
  for (int ii = o0; ii < o1; ++ii) {
    int e = order[ii];
    const float* ear  = &eas[e * EDIM];
    const float* hsrc = &hs[sl[e] * HID + fbase];
    float ef[64];
#pragma unroll
    for (int f = 0; f < 64; ++f) ef[f] = bes[fbase + f];
    for (int k = 0; k < EDIM; ++k) {
      float a = ear[k];
      const float* wr = &Wes[k * HID + fbase];
#pragma unroll
      for (int f = 0; f < 64; ++f) ef[f] = fmaf(a, wr[f], ef[f]);
    }
#pragma unroll
    for (int f = 0; f < 64; ++f) {
      float m = hsrc[f] + ef[f];
      acc[f] += m > 0.f ? m : 0.f;
    }
  }

  float* zout = &z[(size_t)(g * NP + node) * HID + fbase];
#pragma unroll
  for (int f = 0; f < 64; ++f) zout[f] = acc[f];
}

// =====================================================================
// global attention pooling, one block per graph (segment softmax of
// sigmoid gate, weighted sum of h)
// =====================================================================
__global__ void pool_kernel(const float* __restrict__ h,
                            const float* __restrict__ Wg,
                            const float* __restrict__ bg,
                            float* __restrict__ gout) {
  __shared__ float wgs[HID];
  __shared__ float gate[NP];
  __shared__ float exv[NP];
  __shared__ float red[2];           // [0]=max, [1]=1/sum

  const int g = blockIdx.x;
  const int tid = threadIdx.x;
  if (tid < HID) wgs[tid] = Wg[tid];
  __syncthreads();

  if (tid < NP) {
    const float* hr = &h[(size_t)(g * NP + tid) * HID];
    float d = 0.f;
    for (int c = 0; c < HID; ++c) d += hr[c] * wgs[c];
    d += bg[0];
    gate[tid] = 1.f / (1.f + expf(-d));     // sigmoid
  }
  __syncthreads();
  if (tid == 0) {
    float m = gate[0];
    for (int i = 1; i < NP; ++i) m = fmaxf(m, gate[i]);
    red[0] = m;
  }
  __syncthreads();
  if (tid < NP) exv[tid] = expf(gate[tid] - red[0]);
  __syncthreads();
  if (tid == 0) {
    float s = 0.f;
    for (int i = 0; i < NP; ++i) s += exv[i];
    red[1] = 1.f / s;
  }
  __syncthreads();

  // tid = feature
  float a = 0.f;
  for (int n = 0; n < NP; ++n)
    a += exv[n] * h[(size_t)(g * NP + n) * HID + tid];
  gout[(size_t)g * HID + tid] = a * red[1];
}

// =====================================================================
// kernel_launch
// =====================================================================
extern "C" void kernel_launch(void* const* d_in, const int* in_sizes, int n_in,
                              void* d_out, int out_size, void* d_ws, size_t ws_size,
                              hipStream_t stream) {
  (void)in_sizes; (void)n_in; (void)out_size; (void)ws_size;

  const float* x         = (const float*)d_in[0];
  const float* edge_attr = (const float*)d_in[1];
  const float* W_init    = (const float*)d_in[2];
  const float* b_init    = (const float*)d_in[3];
  const float* bn0_g     = (const float*)d_in[4];
  const float* bn0_b     = (const float*)d_in[5];
  const float* eps_l     = (const float*)d_in[6];
  const float* We        = (const float*)d_in[7];
  const float* be        = (const float*)d_in[8];
  const float* Wc1       = (const float*)d_in[9];
  const float* bc1       = (const float*)d_in[10];
  const float* Wc2       = (const float*)d_in[11];
  const float* bc2       = (const float*)d_in[12];
  const float* bn_g      = (const float*)d_in[13];
  const float* bn_b      = (const float*)d_in[14];
  const float* Wg        = (const float*)d_in[15];
  const float* bg        = (const float*)d_in[16];
  const float* Wo        = (const float*)d_in[17];
  const float* bo        = (const float*)d_in[18];
  const int*   edge_index= (const int*)d_in[19];
  float* out             = (float*)d_out;

  // ---- workspace carve-up (all 256B aligned) ----
  char* w = (char*)d_ws;
  auto take = [&](size_t bytes) -> char* {
    char* p = w;
    w += (bytes + 255) & ~(size_t)255;
    return p;
  };
  float* pos  = (float*)take((size_t)NN * PEK * 4);
  float* bufA = (float*)take((size_t)NN * HID * 4);   // h / residual
  float* bufB = (float*)take((size_t)NN * HID * 4);   // z / pre-BN sum
  float* bufC = (float*)take((size_t)NN * HID * 4);   // conv1 output
  float* gbuf = (float*)take((size_t)NG * HID * 4);   // pooled graphs
  float* mu   = (float*)take(HID * 4);
  float* rs   = (float*)take(HID * 4);
  unsigned short* WbInit = (unsigned short*)take((size_t)HID * KINIT * 2);
  unsigned short* WbC    = (unsigned short*)take((size_t)8 * HID * HID * 2);
  unsigned short* WbO    = (unsigned short*)take((size_t)HID * HID * 2);

  // ---- weight conversion (f32 -> bf16, transposed [n][k]) ----
  cvt_init_kernel<<<(HID * KINIT + 255) / 256, 256, 0, stream>>>(W_init, WbInit);
  cvt_wc_kernel<<<(8 * HID * HID + 255) / 256, 256, 0, stream>>>(Wc1, Wc2, WbC);
  cvt_wo_kernel<<<(HID * HID + 255) / 256, 256, 0, stream>>>(Wo, WbO);

  // ---- Laplacian positional encodings ----
  {
    size_t lds = (size_t)(2 * 64 * 64 + 64 + 64 + 32 + 32) * 4 +
                 (size_t)(32 + 32 + 64) * 4;
    lap_pe_kernel<<<NG, NP, lds, stream>>>(edge_index, pos);
  }

  auto gemm = [&](const float* A, int lda, int kaCols, const float* A2,
                  const unsigned short* Bt, int Kp, int M,
                  const float* bias, const float* resid, float* C, int act) {
    size_t lds = (size_t)512 * Kp + (size_t)64 * ASTR * 2;
    wmma_gemm256_kernel<<<M / 64, 256, lds, stream>>>(
        A, lda, kaCols, A2, Bt, Kp, bias, resid, C, act);
  };

  // ---- init: h = BN0( lrelu( [x,pos] @ W_init + b_init ) ) ----
  gemm(x, IND, IND + PEK, pos, WbInit, KINIT, NN, b_init, nullptr, bufB, /*lrelu*/2);
  bn_stats_kernel<<<HID, 256, 0, stream>>>(bufB, NN, mu, rs);
  bn_norm_kernel<<<(NN * HID + 255) / 256, 256, 0, stream>>>(
      bufB, mu, rs, bn0_g, bn0_b, bufA, NN * HID, /*no act*/0);

  // ---- 4 GINE layers ----
  size_t gine_lds = (size_t)(NP * HID + EPG * EDIM + EDIM * HID + HID) * 4 +
                    (size_t)(EPG + EPG + 64 + 65 + EPG) * 4;
  for (int l = 0; l < 4; ++l) {
    gine_agg_kernel<<<NG, 256, gine_lds, stream>>>(
        bufA, edge_attr, edge_index,
        We + (size_t)l * EDIM * HID, be + (size_t)l * HID,
        eps_l + l, bufB);
    // conv1 (center tap): relu(z @ W1^T + bc1)
    gemm(bufB, HID, HID, nullptr, WbC + (size_t)l * HID * HID, HID, NN,
         bc1 + (size_t)l * HID, nullptr, bufC, /*relu*/1);
    // conv2 + residual: s = t @ W2^T + bc2 + res
    gemm(bufC, HID, HID, nullptr, WbC + (size_t)(4 + l) * HID * HID, HID, NN,
         bc2 + (size_t)l * HID, bufA, bufB, /*none*/0);
    // h = lrelu(BN(s))
    bn_stats_kernel<<<HID, 256, 0, stream>>>(bufB, NN, mu, rs);
    bn_norm_kernel<<<(NN * HID + 255) / 256, 256, 0, stream>>>(
        bufB, mu, rs, bn_g + (size_t)l * HID, bn_b + (size_t)l * HID,
        bufA, NN * HID, /*lrelu*/2);
  }

  // ---- attention pooling + output projection ----
  pool_kernel<<<NG, 256, 0, stream>>>(bufA, Wg, bg, gbuf);
  gemm(gbuf, HID, HID, nullptr, WbO, HID, NG, bo, nullptr, out, /*none*/0);
}
